// SupervisedContrastiveLoss_19164144075503
// MI455X (gfx1250) — matrix operand order repaired and checked by reading, not compiled, not run on hardware
//
#include <hip/hip_runtime.h>
#include <math.h>

typedef __attribute__((ext_vector_type(16))) _Float16 v16h;
typedef __attribute__((ext_vector_type(8)))  _Float16 v8h;
typedef __attribute__((ext_vector_type(8)))  float    v8f;

#define MAX_NEG   64
#define TEMP_POS  5.0f
#define TEMP_NEG  10.0f
#define MAX_N     1024   // reference N is 1024

// ---------------------------------------------------------------------------
// Kernel 1: L2-normalize each row, emit f16 copy for the WMMA GEMM.
// ---------------------------------------------------------------------------
__global__ void __launch_bounds__(256)
normalize_rows_f16(const float* __restrict__ x, _Float16* __restrict__ h16,
                   int n, int d) {
  const int row = blockIdx.x;
  const float* xr = x + (size_t)row * d;
  float ss = 0.0f;
  for (int k = threadIdx.x; k < d; k += blockDim.x) {
    float v = xr[k];
    ss += v * v;
  }
  __shared__ float red[256];
  red[threadIdx.x] = ss;
  __syncthreads();
  for (int s = 128; s > 0; s >>= 1) {
    if (threadIdx.x < s) red[threadIdx.x] += red[threadIdx.x + s];
    __syncthreads();
  }
  const float inv = 1.0f / sqrtf(red[0]);
  _Float16* hr = h16 + (size_t)row * d;
  for (int k = threadIdx.x; k < d; k += blockDim.x)
    hr[k] = (_Float16)(xr[k] * inv);
}

// ---------------------------------------------------------------------------
// Kernel 2: sim = H * H^T via v_wmma_f32_16x16x32_f16.
// One wave per 16x16 output tile; both operands stream from row-major H.
//   A (16x32): lane m=L%16; half=L/16; k runs {c+half*8 .. +7} and {c+16+half*8 .. +7}
//   B (32x16): lane n=L%16; half=L/16; k runs {c+half*8 .. +7} and {c+16+half*8 .. +7}
//   C/D:       lane n=L%16; VGPR r -> m = 8*half + r
// ---------------------------------------------------------------------------
__global__ void __launch_bounds__(256)
sim_wmma(const _Float16* __restrict__ h16, float* __restrict__ sim,
         int n, int d) {
  const int lane  = threadIdx.x & 31;
  const int wid   = blockIdx.x * (blockDim.x >> 5) + (threadIdx.x >> 5);
  const int ntile = n >> 4;
  if (wid >= ntile * ntile) return;
  const int ti = wid / ntile;
  const int tj = wid % ntile;
  const int r    = lane & 15;
  const int half = lane >> 4;

  const _Float16* arow = h16 + (size_t)(ti * 16 + r) * d;
  const _Float16* brow = h16 + (size_t)(tj * 16 + r) * d;

  v8f acc = {};
  for (int k0 = 0; k0 < d; k0 += 32) {
    __builtin_prefetch(arow + k0 + 32, 0, 0);   // global_prefetch_b8
    __builtin_prefetch(brow + k0 + 32, 0, 0);
    const v8h a0 = *(const v8h*)(arow + k0 + half * 8);
    const v8h a1 = *(const v8h*)(arow + k0 + 16 + half * 8);
    const v8h b0 = *(const v8h*)(brow + k0 + half * 8);
    const v8h b1 = *(const v8h*)(brow + k0 + 16 + half * 8);
    v16h a, b;
#pragma unroll
    for (int t = 0; t < 8; ++t) {
      a[t] = a0[t]; a[t + 8] = a1[t];
      b[t] = b0[t]; b[t + 8] = b1[t];
    }
    // (neg_a, A, neg_b, B, c_mod, C, reuse_a, reuse_b)
    acc = __builtin_amdgcn_wmma_f32_16x16x32_f16(
        false, a, false, b, (short)0, acc, false, false);
  }

  float* out = sim + (size_t)(ti * 16 + half * 8) * n + (tj * 16 + r);
#pragma unroll
  for (int v = 0; v < 8; ++v) out[(size_t)v * n] = acc[v];
}

// ---------------------------------------------------------------------------
// JAX threefry2x32 (key = random.key(1) -> (0,1)) for the noise matrix,
// matching jax.random.uniform(key(1), (n, n)).
// ---------------------------------------------------------------------------
__device__ __forceinline__ unsigned rotl32(unsigned x, int r) {
  return (x << r) | (x >> (32 - r));
}

__device__ __forceinline__ void threefry2x32_key01(unsigned c0, unsigned c1,
                                                   unsigned* o0, unsigned* o1) {
  const unsigned ks0 = 0u, ks1 = 1u;
  const unsigned ks2 = 0x1BD11BDAu ^ ks0 ^ ks1;
  unsigned x0 = c0 + ks0, x1 = c1 + ks1;
  const int ra[4] = {13, 15, 26, 6};
  const int rb[4] = {17, 29, 16, 24};
#pragma unroll
  for (int t = 0; t < 4; ++t) { x0 += x1; x1 = rotl32(x1, ra[t]); x1 ^= x0; }
  x0 += ks1; x1 += ks2 + 1u;
#pragma unroll
  for (int t = 0; t < 4; ++t) { x0 += x1; x1 = rotl32(x1, rb[t]); x1 ^= x0; }
  x0 += ks2; x1 += ks0 + 2u;
#pragma unroll
  for (int t = 0; t < 4; ++t) { x0 += x1; x1 = rotl32(x1, ra[t]); x1 ^= x0; }
  x0 += ks0; x1 += ks1 + 3u;
#pragma unroll
  for (int t = 0; t < 4; ++t) { x0 += x1; x1 = rotl32(x1, rb[t]); x1 ^= x0; }
  x0 += ks1; x1 += ks2 + 4u;
#pragma unroll
  for (int t = 0; t < 4; ++t) { x0 += x1; x1 = rotl32(x1, ra[t]); x1 ^= x0; }
  x0 += ks2; x1 += ks0 + 5u;
  *o0 = x0; *o1 = x1;
}

// ---------------------------------------------------------------------------
// Kernel 3: one block per row. Generate noise, select top-64 negatives
// (uniform random subset, ties -> lower index like lax.top_k), then
// sum softplus(neg/10 - pos/5) over (pos x neg) pairs.
// ---------------------------------------------------------------------------
__global__ void __launch_bounds__(256)
row_loss_kernel(const float* __restrict__ sim, const int* __restrict__ labels,
                float* __restrict__ row_loss, float* __restrict__ row_has,
                int n) {
  const int i   = blockIdx.x;
  const int tid = threadIdx.x;
  const int li  = labels[i];

  __shared__ float s_score[MAX_N];
  __shared__ float s_rv[256];
  __shared__ int   s_ri[256];
  __shared__ float s_negsim[MAX_NEG];
  __shared__ int   s_negidx[MAX_NEG];
  __shared__ int   s_nneg;

  const unsigned total = (unsigned)n * (unsigned)n;
  const unsigned H = total >> 1;  // JAX splits the iota counter in halves

  for (int j = tid; j < n; j += blockDim.x) {
    float sc = -INFINITY;
    if (labels[j] != li) {
      const unsigned p = (unsigned)i * (unsigned)n + (unsigned)j;
      const unsigned q = (p < H) ? p : (p - H);
      unsigned b0, b1;
      threefry2x32_key01(q, q + H, &b0, &b1);
      const unsigned bits = (p < H) ? b0 : b1;
      sc = __uint_as_float(0x3F800000u | (bits >> 9)) - 1.0f;  // [0,1)
    }
    s_score[j] = sc;
  }
  if (tid == 0) s_nneg = 0;
  __syncthreads();

  // 64 rounds of block-wide argmax (tie -> lower index); mark taken with -inf.
  for (int it = 0; it < MAX_NEG; ++it) {
    float best = -INFINITY;
    int   bi   = n;
    for (int j = tid; j < n; j += blockDim.x) {
      float v = s_score[j];
      if (v > best) { best = v; bi = j; }   // ascending j => lower-index ties
    }
    s_rv[tid] = best; s_ri[tid] = bi;
    __syncthreads();
    for (int s = 128; s > 0; s >>= 1) {
      if (tid < s) {
        float ov = s_rv[tid + s]; int oi = s_ri[tid + s];
        if (ov > s_rv[tid] || (ov == s_rv[tid] && oi < s_ri[tid])) {
          s_rv[tid] = ov; s_ri[tid] = oi;
        }
      }
      __syncthreads();
    }
    if (tid == 0) {
      if (s_rv[0] > -INFINITY) {          // fewer than 64 negatives -> stop
        const int idx = s_ri[0];
        s_negidx[s_nneg] = idx;
        s_score[idx] = -INFINITY;
        s_nneg++;
      }
    }
    __syncthreads();
  }

  const int nneg = s_nneg;
  if (tid < nneg)
    s_negsim[tid] = sim[(size_t)i * n + s_negidx[tid]] * (1.0f / TEMP_NEG);
  __syncthreads();

  float acc = 0.0f;
  int   myPos = 0;
  for (int j = tid; j < n; j += blockDim.x) {
    if (j != i && labels[j] == li) {
      myPos++;
      const float pos = sim[(size_t)i * n + j] * (1.0f / TEMP_POS);
      for (int k = 0; k < nneg; ++k) {
        const float x = s_negsim[k] - pos;
        // softplus(x) = logaddexp(x, 0)
        acc += fmaxf(x, 0.0f) + log1pf(expf(-fabsf(x)));
      }
    }
  }
  s_rv[tid] = acc; s_ri[tid] = myPos;
  __syncthreads();
  for (int s = 128; s > 0; s >>= 1) {
    if (tid < s) { s_rv[tid] += s_rv[tid + s]; s_ri[tid] += s_ri[tid + s]; }
    __syncthreads();
  }
  if (tid == 0) {
    const int   npos   = s_ri[0];
    const float npairs = (float)(npos * nneg);
    row_loss[i] = (npos > 0) ? s_rv[0] / fmaxf(npairs, 1.0f) : 0.0f;
    row_has[i]  = (npos > 0) ? 1.0f : 0.0f;
  }
}

// ---------------------------------------------------------------------------
// Kernel 4: final mean over rows that had positives.
// ---------------------------------------------------------------------------
__global__ void __launch_bounds__(256)
final_reduce(const float* __restrict__ row_loss,
             const float* __restrict__ row_has,
             float* __restrict__ out, int n) {
  __shared__ float s0[256], s1[256];
  float a = 0.0f, b = 0.0f;
  for (int j = threadIdx.x; j < n; j += 256) { a += row_loss[j]; b += row_has[j]; }
  s0[threadIdx.x] = a; s1[threadIdx.x] = b;
  __syncthreads();
  for (int s = 128; s > 0; s >>= 1) {
    if (threadIdx.x < s) {
      s0[threadIdx.x] += s0[threadIdx.x + s];
      s1[threadIdx.x] += s1[threadIdx.x + s];
    }
    __syncthreads();
  }
  if (threadIdx.x == 0) {
    const float cnt = s1[0];
    out[0] = (cnt > 0.0f) ? s0[0] / fmaxf(cnt, 1.0f) : 0.0f;
  }
}

// ---------------------------------------------------------------------------
extern "C" void kernel_launch(void* const* d_in, const int* in_sizes, int n_in,
                              void* d_out, int out_size, void* d_ws, size_t ws_size,
                              hipStream_t stream) {
  const float* hv     = (const float*)d_in[0];
  const int*   labels = (const int*)d_in[1];
  const int n = in_sizes[1];          // 1024
  const int d = in_sizes[0] / n;      // 512

  char* ws = (char*)d_ws;
  size_t off = 0;
  _Float16* h16 = (_Float16*)(ws + off);
  off += (size_t)n * d * sizeof(_Float16);
  off = (off + 255) & ~(size_t)255;
  float* sim = (float*)(ws + off);
  off += (size_t)n * n * sizeof(float);
  float* row_loss = (float*)(ws + off); off += (size_t)n * sizeof(float);
  float* row_has  = (float*)(ws + off); off += (size_t)n * sizeof(float);

  normalize_rows_f16<<<n, 256, 0, stream>>>(hv, h16, n, d);

  const int ntile = n / 16;
  const int tiles = ntile * ntile;          // one wave per 16x16 tile
  sim_wmma<<<(tiles + 7) / 8, 256, 0, stream>>>(h16, sim, n, d);

  row_loss_kernel<<<n, 256, 0, stream>>>(sim, labels, row_loss, row_has, n);

  final_reduce<<<1, 256, 0, stream>>>(row_loss, row_has, (float*)d_out, n);
}
